// kNN_VC_22625887716023
// MI455X (gfx1250) — matrix operand order repaired and checked by reading, not compiled, not run on hardware
//
#include <hip/hip_runtime.h>
#include <math.h>

// ---------------------------------------------------------------------------
// kNN-VC on gfx1250: cosine-sim kNN via V_WMMA_F32_16X16X32_BF16.
//   A = 16 source rows (M), B = 16 target rows (N), C = f32 sims.
//   C layout => each lane owns ONE target column and 8 source rows, so the
//   per-source running top-4 update is entirely lane-local in the hot loop.
//   Targets are pre-normalized to bf16 ONCE (halves hot-loop HBM traffic and
//   removes all f32->bf16 conversion VALU from the K-loop).
//   A-tile (128 src rows x 1024 bf16) staged to LDS by the Tensor Data Mover
//   (pad_enable gives a 2064B row stride -> conflict-mitigated ds_load_b64).
//   K-loop is 2-deep software pipelined: loads for step kt+2 issue before the
//   WMMAs of step kt, giving each load a full body of in-flight latency.
// ---------------------------------------------------------------------------

#define DIMS        1024     // feature dim
#define CT          256      // targets per chunk (per block.x)
#define BSRC        128      // source rows staged per block (block.y)
#define ROW_BYTES   2064     // LDS row stride: 2048B data + 2x8B TDM pads
#define HALF_BYTES  1032     // second 512 elems start here within a row
#define LDS_MAIN_BYTES (BSRC * ROW_BYTES)   // 264192 B <= 320KB/WGP

typedef __attribute__((ext_vector_type(16))) __bf16   v16bf;
typedef __attribute__((ext_vector_type(8)))  float    v8f;
typedef __attribute__((ext_vector_type(4)))  uint32_t u32x4;
typedef __attribute__((ext_vector_type(4)))  int      i32x4;
typedef __attribute__((ext_vector_type(8)))  int      i32x8;

#if __has_builtin(__builtin_amdgcn_tensor_load_to_lds) && __has_builtin(__builtin_amdgcn_s_wait_tensorcnt)
#define USE_TDM 1
#else
#define USE_TDM 0
#endif

struct Top4 { float v0, v1, v2, v3; int i0, i1, i2, i3; };

__device__ __forceinline__ void top4_init(Top4& t) {
  t.v0 = t.v1 = t.v2 = t.v3 = -INFINITY;
  t.i0 = t.i1 = t.i2 = t.i3 = 0;
}

// Branchless insertion into a descending sorted 4-list: 4 v_cmp + cndmask
// chains instead of EXEC-mask branch nests (lane-divergence friendly).
__device__ __forceinline__ void top4_ins(Top4& t, float v, int i) {
  const bool g0 = v > t.v0, g1 = v > t.v1, g2 = v > t.v2, g3 = v > t.v3;
  const float nv3 = g3 ? (g2 ? t.v2 : v) : t.v3;  const int ni3 = g3 ? (g2 ? t.i2 : i) : t.i3;
  const float nv2 = g2 ? (g1 ? t.v1 : v) : t.v2;  const int ni2 = g2 ? (g1 ? t.i1 : i) : t.i2;
  const float nv1 = g1 ? (g0 ? t.v0 : v) : t.v1;  const int ni1 = g1 ? (g0 ? t.i0 : i) : t.i1;
  const float nv0 = g0 ? v : t.v0;                const int ni0 = g0 ? i : t.i0;
  t.v0 = nv0; t.v1 = nv1; t.v2 = nv2; t.v3 = nv3;
  t.i0 = ni0; t.i1 = ni1; t.i2 = ni2; t.i3 = ni3;
}

// ---- Pass 1: normalize source rows -> bf16 ---------------------------------
__global__ void knn_src_norm(const float* __restrict__ src, __bf16* __restrict__ sbf) {
  const int row = blockIdx.x;
  const int tid = threadIdx.x;                       // 256 threads, 4 elems each
  const float4 x = ((const float4*)(src + (size_t)row * DIMS))[tid];
  float ss = x.x * x.x + x.y * x.y + x.z * x.z + x.w * x.w;
  __shared__ float red[256];
  red[tid] = ss;
  __syncthreads();
  for (int s = 128; s > 0; s >>= 1) {
    if (tid < s) red[tid] += red[tid + s];
    __syncthreads();
  }
  const float rn = rsqrtf(red[0]);
  union { __bf16 h[4]; uint2 u; } o;
  o.h[0] = (__bf16)(x.x * rn); o.h[1] = (__bf16)(x.y * rn);
  o.h[2] = (__bf16)(x.z * rn); o.h[3] = (__bf16)(x.w * rn);
  ((uint2*)(sbf + (size_t)row * DIMS))[tid] = o.u;
}

// ---- Pass 2: normalize target rows -> bf16 (one wave32 per row) ------------
__global__ void knn_tgt_norm_bf16(const float* __restrict__ tgt, __bf16* __restrict__ tbf, int NT) {
  const int gwarp = (int)((blockIdx.x * blockDim.x + threadIdx.x) >> 5);
  const int lane  = threadIdx.x & 31;
  if (gwarp >= NT) return;
  const float4* p = (const float4*)(tgt + (size_t)gwarp * DIMS);
  float4 x[8];
  float ss = 0.f;
#pragma unroll
  for (int i = 0; i < 8; ++i) {                      // 32 lanes * 8 * float4 = 1024
    x[i] = p[lane + 32 * i];
    ss += x[i].x * x[i].x + x[i].y * x[i].y + x[i].z * x[i].z + x[i].w * x[i].w;
  }
#pragma unroll
  for (int m = 16; m > 0; m >>= 1) ss += __shfl_xor(ss, m, 32);  // all lanes get sum
  const float rn = rsqrtf(ss);
  uint2* o = (uint2*)(tbf + (size_t)gwarp * DIMS);
#pragma unroll
  for (int i = 0; i < 8; ++i) {
    union { __bf16 h[4]; uint2 u; } w;
    w.h[0] = (__bf16)(x[i].x * rn); w.h[1] = (__bf16)(x[i].y * rn);
    w.h[2] = (__bf16)(x[i].z * rn); w.h[3] = (__bf16)(x[i].w * rn);
    o[lane + 32 * i] = w.u;
  }
}

#if USE_TDM
// Issue one TDM descriptor: copy BSRC x DIMS bf16 rows (row-major, stride
// DIMS) from global to LDS at ldsOff, padding 2 DWORDs after every 256 DWORDs
// stored (rows are 512 DW -> row stride becomes 516 DW = 2064B).
__device__ __forceinline__ void tdm_stage_A(const __bf16* gsrc, uint32_t ldsOff) {
  const unsigned long long ga = (unsigned long long)(uintptr_t)gsrc;
  u32x4 g0;
  g0[0] = 1u;                                          // count=1, user mode
  g0[1] = ldsOff;                                      // lds_addr (bytes)
  g0[2] = (uint32_t)ga;                                // global_addr[31:0]
  g0[3] = (uint32_t)((ga >> 32) & 0x1FFFFFFu) | (2u << 30);  // [56:32] | type=2
  i32x8 g1;
  g1[0] = (1 << 16)      // data_size = 2 bytes
        | (1 << 20)      // pad_enable
        | (7 << 22)      // pad_interval: every 256 DWORDs
        | (1 << 25);     // pad_amount: 2 DWORDs
  g1[1] = (int)((DIMS & 0xFFFF) << 16);                // tensor_dim0[15:0] @ bits 63:48
  g1[2] = (int)((DIMS >> 16) | ((BSRC & 0xFFFF) << 16)); // dim0 hi | tensor_dim1 lo
  g1[3] = (int)((BSRC >> 16) | ((DIMS & 0xFFFF) << 16)); // dim1 hi | tile_dim0
  g1[4] = (int)(BSRC & 0xFFFF);                        // tile_dim1 | tile_dim2=0
  g1[5] = (int)DIMS;                                   // tensor_dim0_stride[31:0]
  g1[6] = 0;                                           // stride hi | dim1_stride lo
  g1[7] = 0;
  const i32x4 z4 = {0, 0, 0, 0};
#if defined(__clang_major__) && (__clang_major__ >= 23)
  const i32x8 z8 = {0, 0, 0, 0, 0, 0, 0, 0};
  __builtin_amdgcn_tensor_load_to_lds(g0, g1, z4, z4, z8, 0);
#else
  __builtin_amdgcn_tensor_load_to_lds(g0, g1, z4, z4, 0);
#endif
}
#endif

// A-fragment load: 16 bf16 for this lane's (row, k-tile) slice.
// Layout: 2048B of row data with an 8B pad after the first 1024B, so
// byte offset of element K is 2K + (K>=512 ? 8 : 0).  Four ds_load_b64
// (compiler pairs them into ds_load_2addr_b64).
__device__ __forceinline__ v16bf load_a_frag(const char* aRowB, int kb, int hi) {
  const int off = 2 * (kb + hi * 16) + ((kb >= 512) ? 8 : 0);
  union { uint2 u[4]; v16bf v; } A;
  const char* p = aRowB + off;
  A.u[0] = *(const uint2*)(p);        // K+0..3
  A.u[1] = *(const uint2*)(p + 8);    // K+4..7
  A.u[2] = *(const uint2*)(p + 16);   // K+8..11
  A.u[3] = *(const uint2*)(p + 24);   // K+12..15
  return A.v;
}

// B-fragment: 16 consecutive pre-normalized bf16 of one target row.
struct BF { uint4 a, b; };
__device__ __forceinline__ BF load_b_frag(const __bf16* base, int kb) {
  const uint4* f = (const uint4*)(base + kb);
  BF r; r.a = f[0]; r.b = f[1];
  return r;
}
__device__ __forceinline__ v16bf bf_cast(const BF& x) {
  union { uint4 u[2]; v16bf v; } c;
  c.u[0] = x.a; c.u[1] = x.b;
  return c.v;
}

// ---- Pass 3: WMMA similarity + per-chunk top-4 -----------------------------
// grid = (nchunks, NSRC/BSRC), block = 256 (8 waves); dyn LDS = LDS_MAIN_BYTES.
// Requires: chunk target counts are even multiples of 16 (true for NT=100000, CT=256).
__global__ void __launch_bounds__(256, 1)
knn_sim_kernel(const __bf16* __restrict__ sbf, const __bf16* __restrict__ tbf,
               float* __restrict__ pval, int* __restrict__ pidx, int NT, int NS) {
  extern __shared__ char smem[];

  const int tid       = threadIdx.x;
  const int cx        = blockIdx.x;
  const int srcBase   = blockIdx.y * BSRC;
  const int chunkBase = cx * CT;
  const int nt_here   = (NT - chunkBase < CT) ? (NT - chunkBase) : CT;
  const int ntiles    = nt_here >> 4;                // even by construction

  // Stage BSRC normalized-bf16 source rows into LDS (row stride 2064B).
#if USE_TDM
  if (tid < 32) {                                    // wave 0 drives the TDM
    tdm_stage_A(sbf + (size_t)srcBase * DIMS, (uint32_t)(uintptr_t)smem);
    __builtin_amdgcn_s_wait_tensorcnt(0);
  }
#else
  {
    const int row  = tid >> 1;
    const int half = tid & 1;
    const uint2* g = (const uint2*)(sbf + (size_t)(srcBase + row) * DIMS + half * (DIMS / 2));
    uint2* l = (uint2*)(smem + (size_t)row * ROW_BYTES + half * HALF_BYTES);
    for (int i = 0; i < (DIMS / 2) / 4; ++i) l[i] = g[i];
  }
#endif
  __syncthreads();

  const int lane = tid & 31;
  const int warp = tid >> 5;
  const int lrow = lane & 15;        // N-column (target) / A row (source) selector
  const int hi   = lane >> 4;        // lane group: selects K-split halves
  const char* aRowB = smem + (size_t)(warp * 16 + lrow) * ROW_BYTES;

  Top4 st[8];
#pragma unroll
  for (int r = 0; r < 8; ++r) top4_init(st[r]);

  for (int p = 0; p < ntiles; p += 2) {
    const int t0 = chunkBase + p * 16 + lrow;
    const int t1 = t0 + 16;
    const __bf16* tp0 = tbf + (size_t)t0 * DIMS + hi * 16;
    const __bf16* tp1 = tbf + (size_t)t1 * DIMS + hi * 16;

    if (p + 2 < ntiles) {                            // global_prefetch_b8 ahead
      __builtin_prefetch(tp0 + 32 * (size_t)DIMS, 0, 0);
      __builtin_prefetch(tp1 + 32 * (size_t)DIMS, 0, 0);
    }

    v8f acc0 = {0.f, 0.f, 0.f, 0.f, 0.f, 0.f, 0.f, 0.f};
    v8f acc1 = {0.f, 0.f, 0.f, 0.f, 0.f, 0.f, 0.f, 0.f};

    // 2-deep software pipeline: two fragment slots (A/B), each refilled two
    // K-steps ahead of use so every load has a full body in flight.
    v16bf aA  = load_a_frag(aRowB, 0, hi);
    BF    bA0 = load_b_frag(tp0, 0),  bA1 = load_b_frag(tp1, 0);
    v16bf aB  = load_a_frag(aRowB, 32, hi);
    BF    bB0 = load_b_frag(tp0, 32), bB1 = load_b_frag(tp1, 32);

#pragma unroll 2
    for (int kt = 0; kt < DIMS / 32; kt += 2) {
      const int k2 = (kt + 2 < DIMS / 32) ? (kt + 2) * 32 : 0;    // wrap: harmless reload
      const int k3 = (kt + 3 < DIMS / 32) ? (kt + 3) * 32 : 32;
      // even step: issue refill for kt+2, then consume slot A (loaded 2 steps ago)
      const v16bf aN = load_a_frag(aRowB, k2, hi);
      const BF    n0 = load_b_frag(tp0, k2), n1 = load_b_frag(tp1, k2);
      acc0 = __builtin_amdgcn_wmma_f32_16x16x32_bf16(false, aA, false, bf_cast(bA0), (short)0, acc0, false, false);
      acc1 = __builtin_amdgcn_wmma_f32_16x16x32_bf16(false, aA, false, bf_cast(bA1), (short)0, acc1, false, false);
      aA = aN; bA0 = n0; bA1 = n1;
      // odd step: issue refill for kt+3, then consume slot B
      const v16bf aM = load_a_frag(aRowB, k3, hi);
      const BF    m0 = load_b_frag(tp0, k3), m1 = load_b_frag(tp1, k3);
      acc0 = __builtin_amdgcn_wmma_f32_16x16x32_bf16(false, aB, false, bf_cast(bB0), (short)0, acc0, false, false);
      acc1 = __builtin_amdgcn_wmma_f32_16x16x32_bf16(false, aB, false, bf_cast(bB1), (short)0, acc1, false, false);
      aB = aM; bB0 = m0; bB1 = m1;
    }

    // Lane-local top-4 update: acc VGPR r holds source row m = r + hi*8,
    // this lane's target column is t0 / t1.
#pragma unroll
    for (int r = 0; r < 8; ++r) {
      top4_ins(st[r], acc0[r], t0);
      top4_ins(st[r], acc1[r], t1);
    }
  }

  // Merge the 16 target-residue lanes per source row via LDS (reuse A region).
  __syncthreads();
  float* mv = (float*)smem;                  // [8 warps][16 m][16 slots][4]
  int*   mi = (int*)(smem + 8 * 16 * 16 * 4 * sizeof(float));
#pragma unroll
  for (int r = 0; r < 8; ++r) {
    const int mm   = r + hi * 8;
    const int base = ((warp * 16 + mm) * 16 + lrow) * 4;
    mv[base + 0] = st[r].v0; mv[base + 1] = st[r].v1; mv[base + 2] = st[r].v2; mv[base + 3] = st[r].v3;
    mi[base + 0] = st[r].i0; mi[base + 1] = st[r].i1; mi[base + 2] = st[r].i2; mi[base + 3] = st[r].i3;
  }
  __syncthreads();
  if (tid < 128) {
    const int w = tid >> 4, m = tid & 15;
    Top4 f; top4_init(f);
    const int base = (w * 16 + m) * 16 * 4;
    for (int s = 0; s < 64; ++s) top4_ins(f, mv[base + s], mi[base + s]);
    const int srcRow = srcBase + w * 16 + m;
    const size_t o = ((size_t)cx * NS + srcRow) * 4;
    pval[o + 0] = f.v0; pval[o + 1] = f.v1; pval[o + 2] = f.v2; pval[o + 3] = f.v3;
    pidx[o + 0] = f.i0; pidx[o + 1] = f.i1; pidx[o + 2] = f.i2; pidx[o + 3] = f.i3;
  }
}

// ---- Pass 4: merge chunk partials -> global top-4, gather & mean -----------
__global__ void knn_reduce_gather(const float* __restrict__ pval, const int* __restrict__ pidx,
                                  const float* __restrict__ tgt, float* __restrict__ out,
                                  int ncand, int NS) {
  const int src = blockIdx.x;
  const int tid = threadIdx.x;
  __shared__ float lv[256 * 4];
  __shared__ int   li[256 * 4];
  __shared__ float lv2[32 * 4];
  __shared__ int   li2[32 * 4];
  __shared__ int   fin[4];

  Top4 t; top4_init(t);
  for (int c = tid; c < ncand; c += 256) {
    const int cx = c >> 2, j = c & 3;
    const size_t o = ((size_t)cx * NS + src) * 4 + j;
    top4_ins(t, pval[o], pidx[o]);
  }
  lv[tid * 4 + 0] = t.v0; lv[tid * 4 + 1] = t.v1; lv[tid * 4 + 2] = t.v2; lv[tid * 4 + 3] = t.v3;
  li[tid * 4 + 0] = t.i0; li[tid * 4 + 1] = t.i1; li[tid * 4 + 2] = t.i2; li[tid * 4 + 3] = t.i3;
  __syncthreads();
  if (tid < 32) {
    Top4 m; top4_init(m);
    for (int s = 0; s < 32; ++s) top4_ins(m, lv[tid * 32 + s], li[tid * 32 + s]);
    lv2[tid * 4 + 0] = m.v0; lv2[tid * 4 + 1] = m.v1; lv2[tid * 4 + 2] = m.v2; lv2[tid * 4 + 3] = m.v3;
    li2[tid * 4 + 0] = m.i0; li2[tid * 4 + 1] = m.i1; li2[tid * 4 + 2] = m.i2; li2[tid * 4 + 3] = m.i3;
  }
  __syncthreads();
  if (tid == 0) {
    Top4 f; top4_init(f);
    for (int s = 0; s < 128; ++s) top4_ins(f, lv2[s], li2[s]);
    fin[0] = f.i0; fin[1] = f.i1; fin[2] = f.i2; fin[3] = f.i3;
  }
  __syncthreads();
  const float4 a = ((const float4*)(tgt + (size_t)fin[0] * DIMS))[tid];
  const float4 b = ((const float4*)(tgt + (size_t)fin[1] * DIMS))[tid];
  const float4 c = ((const float4*)(tgt + (size_t)fin[2] * DIMS))[tid];
  const float4 d = ((const float4*)(tgt + (size_t)fin[3] * DIMS))[tid];
  float4 r;
  r.x = (a.x + b.x + c.x + d.x) * 0.25f;
  r.y = (a.y + b.y + c.y + d.y) * 0.25f;
  r.z = (a.z + b.z + c.z + d.z) * 0.25f;
  r.w = (a.w + b.w + c.w + d.w) * 0.25f;
  ((float4*)(out + (size_t)src * DIMS))[tid] = r;
}

// ---------------------------------------------------------------------------
static inline size_t align256(size_t x) { return (x + 255) & ~(size_t)255; }

extern "C" void kernel_launch(void* const* d_in, const int* in_sizes, int n_in,
                              void* d_out, int out_size, void* d_ws, size_t ws_size,
                              hipStream_t stream) {
  const float* src = (const float*)d_in[0];
  const float* tgt = (const float*)d_in[1];
  const int NS = in_sizes[0] / DIMS;                 // 1024
  const int NT = in_sizes[1] / DIMS;                 // 100000
  const int nchunks = (NT + CT - 1) / CT;            // 391

  char* ws = (char*)d_ws;
  __bf16* sbf = (__bf16*)ws;              ws += align256((size_t)NS * DIMS * sizeof(__bf16));
  __bf16* tbf = (__bf16*)ws;              ws += align256((size_t)NT * DIMS * sizeof(__bf16));
  float*  pval = (float*)ws;              ws += align256((size_t)nchunks * NS * 4 * sizeof(float));
  int*    pidx = (int*)ws;

  (void)hipFuncSetAttribute(reinterpret_cast<const void*>(&knn_sim_kernel),
                            hipFuncAttributeMaxDynamicSharedMemorySize, LDS_MAIN_BYTES);

  knn_src_norm<<<NS, 256, 0, stream>>>(src, sbf);
  knn_tgt_norm_bf16<<<(NT * 32 + 255) / 256, 256, 0, stream>>>(tgt, tbf, NT);
  dim3 grid(nchunks, NS / BSRC);
  knn_sim_kernel<<<grid, 256, LDS_MAIN_BYTES, stream>>>(sbf, tbf, pval, pidx, NT, NS);
  knn_reduce_gather<<<NS, 256, 0, stream>>>(pval, pidx, tgt, (float*)d_out, nchunks * 4, NS);
}
